// _MultiHeadAttention_7172595384540
// MI455X (gfx1250) — matrix operand order repaired and checked by reading, not compiled, not run on hardware
//
#include <hip/hip_runtime.h>

#define BS     2
#define TSEQ   2048
#define DMODEL 1024
#define NH     16
#define DH     64
#define SCALE  0.125f   // 1/sqrt(64)

typedef __bf16 bf16_t;
typedef __attribute__((ext_vector_type(16))) __bf16 v16bf;
typedef __attribute__((ext_vector_type(8)))  __bf16 v8bf;
typedef __attribute__((ext_vector_type(4)))  __bf16 v4bf;
typedef __attribute__((ext_vector_type(8)))  float  v8f;
typedef __attribute__((ext_vector_type(4)))  float  v4f;

// ---- CDNA5 async copy: global -> LDS, tracked by ASYNCcnt ------------------
#define WAIT_ASYNC(n) asm volatile("s_wait_asynccnt %0" :: "n"(n) : "memory")

static __device__ inline void async_b128(void* lds, const void* gaddr) {
  // flat LDS aperture: addr[31:0] is the wave-relative LDS byte offset
  unsigned off = (unsigned)(size_t)lds;
  asm volatile("global_load_async_to_lds_b128 %0, %1, off"
               :: "v"(off), "v"(gaddr) : "memory");
}

static __device__ inline v8f zero8() {
  v8f z; for (int i = 0; i < 8; ++i) z[i] = 0.f; return z;
}

static __device__ inline v8f wmma_bf16(v16bf a, v16bf b, v8f c) {
  return __builtin_amdgcn_wmma_f32_16x16x32_bf16(
      false, a, false, b, (short)0, c, false, false);
}

// A-fragment (16x32): lane holds row l16; k-groups [half*8,+8) and [16+half*8,+8)
static __device__ inline v16bf frag_a(const bf16_t* base, int ld, int kOff, int l16, int half) {
  const v8bf lo = *(const v8bf*)(base + (size_t)l16 * ld + kOff + half * 8);
  const v8bf hi = *(const v8bf*)(base + (size_t)l16 * ld + kOff + 16 + half * 8);
  return __builtin_shufflevector(lo, hi, 0,1,2,3,4,5,6,7,8,9,10,11,12,13,14,15);
}

// B-fragment (32x16) from N-major rows: lane holds col l16; k-groups [half*8,+8) and [16+half*8,+8)
static __device__ inline v16bf frag_b(const bf16_t* base, int ld, int kOff, int l16, int half) {
  const v8bf lo = *(const v8bf*)(base + (size_t)l16 * ld + kOff + half * 8);
  const v8bf hi = *(const v8bf*)(base + (size_t)l16 * ld + kOff + 16 + half * 8);
  return __builtin_shufflevector(lo, hi, 0,1,2,3,4,5,6,7,8,9,10,11,12,13,14,15);
}

// ---------------------------------------------------------------------------
// f32 -> bf16 pre-conversion (bandwidth-bound, native v_cvt_pk_bf16_f32)
// ---------------------------------------------------------------------------
__global__ __launch_bounds__(256) void cvt_kernel(
    const float* __restrict__ in, bf16_t* __restrict__ out, int n4)
{
  int i = blockIdx.x * blockDim.x + threadIdx.x;
  int stride = gridDim.x * blockDim.x;
  for (; i < n4; i += stride) {
    v4f v = *(const v4f*)(in + (size_t)i * 4);
    *(v4bf*)(out + (size_t)i * 4) = __builtin_convertvector(v, v4bf);
  }
}

// ---------------------------------------------------------------------------
// bf16 GEMM: Y = A[M,K] * B[N,K]^T. Block 256 thr = 8 waves, tile 128x128xK32,
// wave -> 32x64 (8 WMMA / K-step). Double-buffered LDS fed by async-to-LDS.
// mode 0: bf16 Y[((b*NH+h)*TSEQ+t)*DH+d]   (head-major: Q, K)
// mode 1: bf16 Y[((b*NH+h)*DH+d)*TSEQ+t]   (per-head transposed: V)
// mode 2: f32  Y[m*DMODEL+n]               (final projection)
// ---------------------------------------------------------------------------
#define MT  128
#define NT_ 128
#define KT  32
#define LDK (KT + 8)            // 40 elems = 80B rows (16B-aligned groups)
#define KSTEPS (DMODEL / KT)    // 32

__global__ __launch_bounds__(256) void gemm_bf16_kernel(
    const bf16_t* __restrict__ A, const bf16_t* __restrict__ B,
    void* __restrict__ outp, int mode)
{
  __shared__ alignas(16) bf16_t sA[2][MT][LDK];   // 2 x 10KB
  __shared__ alignas(16) bf16_t sB[2][NT_][LDK];  // 2 x 10KB

  const int tid  = threadIdx.x;
  const int lane = tid & 31, wid = tid >> 5;
  const int l16  = lane & 15, half = (lane >> 4) & 1;
  const int wm   = wid & 3,  wn = wid >> 2;       // 4 x 2 wave grid
  const int mBlk = blockIdx.y * MT;
  const int nBlk = blockIdx.x * NT_;

  // async staging: 16B chunks; A and B tiles are each 128x32 -> 2 chunks/thread
  const int rA = tid >> 2;
  const int cA = (tid & 3) * 8;

  auto stage = [&](int buf, int k0) {
    async_b128(&sA[buf][rA][cA],      &A[(size_t)(mBlk + rA) * DMODEL + k0 + cA]);
    async_b128(&sA[buf][rA + 64][cA], &A[(size_t)(mBlk + rA + 64) * DMODEL + k0 + cA]);
    async_b128(&sB[buf][rA][cA],      &B[(size_t)(nBlk + rA) * DMODEL + k0 + cA]);
    async_b128(&sB[buf][rA + 64][cA], &B[(size_t)(nBlk + rA + 64) * DMODEL + k0 + cA]);
  };

  v8f acc[2][4];
  for (int i = 0; i < 2; ++i) for (int j = 0; j < 4; ++j) acc[i][j] = zero8();

  stage(0, 0);
  for (int kt = 0; kt < KSTEPS; ++kt) {
    const int cur = kt & 1;
    if (kt + 1 < KSTEPS) {
      stage(cur ^ 1, (kt + 1) * KT);
      WAIT_ASYNC(4);            // in-order completion: previous tile's 4 copies done
    } else {
      WAIT_ASYNC(0);
    }
    __syncthreads();

    v16bf a0 = frag_a(&sA[cur][wm * 32][0],      LDK, 0, l16, half);
    v16bf a1 = frag_a(&sA[cur][wm * 32 + 16][0], LDK, 0, l16, half);
    v16bf b0 = frag_b(&sB[cur][wn * 64][0],      LDK, 0, l16, half);
    v16bf b1 = frag_b(&sB[cur][wn * 64 + 16][0], LDK, 0, l16, half);
    v16bf b2 = frag_b(&sB[cur][wn * 64 + 32][0], LDK, 0, l16, half);
    v16bf b3 = frag_b(&sB[cur][wn * 64 + 48][0], LDK, 0, l16, half);
    acc[0][0] = wmma_bf16(a0, b0, acc[0][0]);
    acc[0][1] = wmma_bf16(a0, b1, acc[0][1]);
    acc[0][2] = wmma_bf16(a0, b2, acc[0][2]);
    acc[0][3] = wmma_bf16(a0, b3, acc[0][3]);
    acc[1][0] = wmma_bf16(a1, b0, acc[1][0]);
    acc[1][1] = wmma_bf16(a1, b1, acc[1][1]);
    acc[1][2] = wmma_bf16(a1, b2, acc[1][2]);
    acc[1][3] = wmma_bf16(a1, b3, acc[1][3]);
    __syncthreads();            // all waves done reading `cur` before it is restaged
  }

  if (mode == 2) {
    float* out = (float*)outp;
    for (int i = 0; i < 2; ++i)
      for (int j = 0; j < 4; ++j) {
        int n = nBlk + wn * 64 + j * 16 + l16;
        for (int r = 0; r < 8; ++r) {
          int m = mBlk + wm * 32 + i * 16 + half * 8 + r;
          out[(size_t)m * DMODEL + n] = acc[i][j][r];
        }
      }
  } else {
    bf16_t* Y = (bf16_t*)outp;
    for (int i = 0; i < 2; ++i)
      for (int j = 0; j < 4; ++j) {
        int nloc = nBlk + wn * 64 + j * 16 + l16;
        int hh = nloc >> 6, dd = nloc & 63;
        for (int r = 0; r < 8; ++r) {
          int m  = mBlk + wm * 32 + i * 16 + half * 8 + r;
          int bI = m >> 11;
          int t  = m & (TSEQ - 1);
          size_t off = (mode == 0)
              ? (((size_t)(bI * NH + hh) * TSEQ + t) * DH + dd)
              : (((size_t)(bI * NH + hh) * DH + dd) * TSEQ + t);
          Y[off] = (bf16_t)acc[i][j][r];
        }
      }
  }
}

// ---------------------------------------------------------------------------
// Flash attention: block = 8 waves; wave owns 16 q-rows; KV tile = 64.
// K/V/Q tiles staged with async-to-LDS; online softmax wave-local.
// ---------------------------------------------------------------------------
#define QT  128
#define JT  64
#define LDQ (DH + 8)   // 72 elems = 144B rows (16B multiple)
#define LDJ (JT + 8)

__global__ __launch_bounds__(256) void attn_kernel(
    const bf16_t* __restrict__ Qb, const bf16_t* __restrict__ Kb,
    const bf16_t* __restrict__ Vt, bf16_t* __restrict__ Ob)
{
  __shared__ alignas(16) bf16_t sQ[QT][LDQ];   // [q][d]  18KB
  __shared__ alignas(16) bf16_t sK[JT][LDQ];   // [j][d]   9KB
  __shared__ alignas(16) bf16_t sV[DH][LDJ];   // [d][j]   9KB
  __shared__ alignas(16) bf16_t sP[QT][LDJ];   // [q][j]  18KB

  const int tid  = threadIdx.x;
  const int lane = tid & 31, wid = tid >> 5;
  const int l16  = lane & 15, half = (lane >> 4) & 1;
  const int qBlk = blockIdx.x * QT;
  const int h    = blockIdx.y, b = blockIdx.z;

  const size_t head = ((size_t)(b * NH + h)) * TSEQ * DH;
  const bf16_t* Qh = Qb + head;                // [t][64]
  const bf16_t* Kh = Kb + head;                // [t][64]
  const bf16_t* Vh = Vt + head;                // [64][t]

  const int rr = tid >> 2;            // 64 rows / pass
  const int cc = (tid & 3) * 8;       // 16B chunk

  // stage Q once (128x64)
  for (int p = 0; p < 2; ++p) {
    async_b128(&sQ[rr + p * 64][cc], &Qh[(size_t)(qBlk + rr + p * 64) * DH + cc]);
  }

  v8f accO[4];
  for (int i = 0; i < 4; ++i) accO[i] = zero8();
  float m_i[8], l_i[8];
  for (int r = 0; r < 8; ++r) { m_i[r] = -1e30f; l_i[r] = 0.f; }

  const int jEnd = qBlk + QT;                  // causal bound
  for (int jBase = 0; jBase < jEnd; jBase += JT) {
    __syncthreads();                           // previous tile fully consumed
    async_b128(&sK[rr][cc], &Kh[(size_t)(jBase + rr) * DH + cc]);
    async_b128(&sV[rr][cc], &Vh[(size_t)rr * TSEQ + jBase + cc]);
    WAIT_ASYNC(0);
    __syncthreads();

    // S = Q K^T  (16 x 64 per wave)
    v8f sf[4];
    const bf16_t* qrow = &sQ[wid * 16][0];
    for (int nt = 0; nt < 4; ++nt) {
      v8f a = zero8();
      a = wmma_bf16(frag_a(qrow, LDQ, 0,  l16, half), frag_b(&sK[nt * 16][0], LDQ, 0,  l16, half), a);
      a = wmma_bf16(frag_a(qrow, LDQ, 32, l16, half), frag_b(&sK[nt * 16][0], LDQ, 32, l16, half), a);
      sf[nt] = a;
    }

    // scale + causal mask (analytic)
    const int qr0 = qBlk + wid * 16 + half * 8;
    for (int nt = 0; nt < 4; ++nt) {
      int jj = jBase + nt * 16 + l16;
      for (int r = 0; r < 8; ++r) {
        float s = sf[nt][r] * SCALE;
        sf[nt][r] = (jj <= (qr0 + r)) ? s : -1e30f;
      }
    }

    // online softmax; row stats live in the 16-lane half
    float rmax[8];
    for (int r = 0; r < 8; ++r)
      rmax[r] = fmaxf(fmaxf(sf[0][r], sf[1][r]), fmaxf(sf[2][r], sf[3][r]));
    for (int off = 1; off < 16; off <<= 1)
      for (int r = 0; r < 8; ++r)
        rmax[r] = fmaxf(rmax[r], __shfl_xor(rmax[r], off, 32));

    float corr[8];
    for (int r = 0; r < 8; ++r) {
      float mn = fmaxf(m_i[r], rmax[r]);
      corr[r] = __expf(m_i[r] - mn);
      m_i[r]  = mn;
      l_i[r] *= corr[r];
    }
    for (int nt = 0; nt < 4; ++nt)
      for (int r = 0; r < 8; ++r)
        accO[nt][r] *= corr[r];

    float rsum[8];
    for (int r = 0; r < 8; ++r) rsum[r] = 0.f;
    for (int nt = 0; nt < 4; ++nt)
      for (int r = 0; r < 8; ++r) {
        float p = __expf(sf[nt][r] - m_i[r]);
        rsum[r] += p;
        sP[wid * 16 + half * 8 + r][nt * 16 + l16] = (bf16_t)p;   // C-frag -> LDS
      }
    for (int off = 1; off < 16; off <<= 1)
      for (int r = 0; r < 8; ++r)
        rsum[r] += __shfl_xor(rsum[r], off, 32);
    for (int r = 0; r < 8; ++r) l_i[r] += rsum[r];

    // O += P V  (wave-local LDS round-trip; DS ops in-order per wave)
    v16bf pa0 = frag_a(&sP[wid * 16][0], LDJ, 0,  l16, half);
    v16bf pa1 = frag_a(&sP[wid * 16][0], LDJ, 32, l16, half);
    for (int nt = 0; nt < 4; ++nt) {
      v8f a = accO[nt];
      a = wmma_bf16(pa0, frag_b(&sV[nt * 16][0], LDJ, 0,  l16, half), a);
      a = wmma_bf16(pa1, frag_b(&sV[nt * 16][0], LDJ, 32, l16, half), a);
      accO[nt] = a;
    }
  }

  // epilogue: O /= l, store bf16 [b][t][h*64+d]
  for (int r = 0; r < 8; ++r) {
    float inv = 1.0f / l_i[r];
    int tq = qBlk + wid * 16 + half * 8 + r;
    size_t rowOff = ((size_t)(b * TSEQ + tq)) * DMODEL + h * DH;
    for (int nt = 0; nt < 4; ++nt)
      Ob[rowOff + nt * 16 + l16] = (bf16_t)(accO[nt][r] * inv);
  }
}

// ---------------------------------------------------------------------------
extern "C" void kernel_launch(void* const* d_in, const int* in_sizes, int n_in,
                              void* d_out, int out_size, void* d_ws, size_t ws_size,
                              hipStream_t stream) {
  const float* q  = (const float*)d_in[0];
  const float* kv = (const float*)d_in[1];
  // d_in[2] = causal mask (computed analytically in-kernel)
  const float* Wq = (const float*)d_in[3];
  const float* Wk = (const float*)d_in[4];
  const float* Wv = (const float*)d_in[5];
  const float* Wo = (const float*)d_in[6];
  float* out = (float*)d_out;

  const size_t actElems = (size_t)BS * TSEQ * DMODEL;   // 4M
  const size_t wElems   = (size_t)DMODEL * DMODEL;      // 1M

  bf16_t* qbf  = (bf16_t*)d_ws;
  bf16_t* kvbf = qbf  + actElems;
  bf16_t* wqb  = kvbf + actElems;
  bf16_t* wkb  = wqb  + wElems;
  bf16_t* wvb  = wkb  + wElems;
  bf16_t* wob  = wvb  + wElems;
  bf16_t* Qb   = wob  + wElems;
  bf16_t* Kb   = Qb   + actElems;
  bf16_t* Vt   = Kb   + actElems;
  bf16_t* Ob   = Vt   + actElems;

  dim3 blk(256);
  // f32 -> bf16 pre-pass (pure bandwidth)
  cvt_kernel<<<dim3((int)(actElems / 4 / 256)), blk, 0, stream>>>(q,  qbf,  (int)(actElems / 4));
  cvt_kernel<<<dim3((int)(actElems / 4 / 256)), blk, 0, stream>>>(kv, kvbf, (int)(actElems / 4));
  cvt_kernel<<<dim3((int)(wElems / 4 / 256)),   blk, 0, stream>>>(Wq, wqb,  (int)(wElems / 4));
  cvt_kernel<<<dim3((int)(wElems / 4 / 256)),   blk, 0, stream>>>(Wk, wkb,  (int)(wElems / 4));
  cvt_kernel<<<dim3((int)(wElems / 4 / 256)),   blk, 0, stream>>>(Wv, wvb,  (int)(wElems / 4));
  cvt_kernel<<<dim3((int)(wElems / 4 / 256)),   blk, 0, stream>>>(Wo, wob,  (int)(wElems / 4));

  dim3 gG(DMODEL / NT_, (BS * TSEQ) / MT);              // (8, 32)
  gemm_bf16_kernel<<<gG, blk, 0, stream>>>(qbf,  wqb, Qb, 0);
  gemm_bf16_kernel<<<gG, blk, 0, stream>>>(kvbf, wkb, Kb, 0);
  gemm_bf16_kernel<<<gG, blk, 0, stream>>>(kvbf, wvb, Vt, 1);

  attn_kernel<<<dim3(TSEQ / QT, NH, BS), blk, 0, stream>>>(Qb, Kb, Vt, Ob);

  gemm_bf16_kernel<<<gG, blk, 0, stream>>>(Ob, wob, out, 2);
}
